// XFeat_35399120453927
// MI455X (gfx1250) — matrix operand order repaired and checked by reading, not compiled, not run on hardware
//
#include <hip/hip_runtime.h>
#include <hip/hip_bf16.h>

typedef __attribute__((ext_vector_type(16))) _Float16 v16h;
typedef __attribute__((ext_vector_type(8)))  _Float16 v8h;
typedef __attribute__((ext_vector_type(8)))  float    v8f;
typedef __attribute__((ext_vector_type(4)))  int      v4i;
typedef __attribute__((address_space(3)))    v4i*     lds_v4i_p;

#define BN_EPS 1e-5f

#if defined(__has_builtin)
#if __has_builtin(__builtin_amdgcn_global_load_async_to_lds_b128)
#define HAVE_ASYNC_LDS 1
#endif
#if __has_builtin(__builtin_amdgcn_s_wait_asynccnt)
#define HAVE_WAIT_ASYNC 1
#endif
#endif
#ifndef HAVE_ASYNC_LDS
#define HAVE_ASYNC_LDS 0
#endif
#ifndef HAVE_WAIT_ASYNC
#define HAVE_WAIT_ASYNC 0
#endif

__device__ __forceinline__ void wait_async_all() {
#if HAVE_WAIT_ASYNC
  __builtin_amdgcn_s_wait_asynccnt(0);
#elif HAVE_ASYNC_LDS
  asm volatile("s_wait_asynccnt 0" ::: "memory");
#endif
}

// ---------------------------------------------------------------- helpers
__device__ __forceinline__ float bsample(const float* p, int ih, int iw,
                                         int oy, int ox, int oh, int ow) {
  float sy = ((oy + 0.5f) * ih) / oh - 0.5f;
  float sx = ((ox + 0.5f) * iw) / ow - 0.5f;
  int y0 = (int)floorf(sy), x0 = (int)floorf(sx);
  float fy = sy - y0, fx = sx - x0;
  int y1 = y0 + 1, x1 = x0 + 1;
  y0 = min(max(y0, 0), ih - 1); y1 = min(max(y1, 0), ih - 1);
  x0 = min(max(x0, 0), iw - 1); x1 = min(max(x1, 0), iw - 1);
  float a = p[(long)y0 * iw + x0], b = p[(long)y0 * iw + x1];
  float c = p[(long)y1 * iw + x0], d = p[(long)y1 * iw + x1];
  return a * (1.f - fy) * (1.f - fx) + b * (1.f - fy) * fx +
         c * fy * (1.f - fx) + d * fy * fx;
}

__device__ __forceinline__ unsigned ford(float f) {
  unsigned u = __float_as_uint(f);
  return (u & 0x80000000u) ? ~u : (u | 0x80000000u);
}
__device__ __forceinline__ float fordinv(unsigned u) {
  u = (u & 0x80000000u) ? (u ^ 0x80000000u) : ~u;
  return __uint_as_float(u);
}

// ---------------------------------------------------------------- gray + instance norm
__global__ void gray_kernel(const float* __restrict__ img, float* __restrict__ xn,
                            float* __restrict__ part) {
  const long HW = 1048576;
  int b = blockIdx.y;
  long i = (long)blockIdx.x * 256 + threadIdx.x;
  float g = (img[((long)b * 3 + 0) * HW + i] + img[((long)b * 3 + 1) * HW + i] +
             img[((long)b * 3 + 2) * HW + i]) * (1.f / 3.f);
  xn[(long)b * HW + i] = g;
  __shared__ float s1[256], s2[256];
  s1[threadIdx.x] = g; s2[threadIdx.x] = g * g;
  __syncthreads();
  for (int o = 128; o > 0; o >>= 1) {
    if (threadIdx.x < (unsigned)o) { s1[threadIdx.x] += s1[threadIdx.x + o]; s2[threadIdx.x] += s2[threadIdx.x + o]; }
    __syncthreads();
  }
  if (threadIdx.x == 0) {
    part[((long)b * 4096 + blockIdx.x) * 2 + 0] = s1[0];
    part[((long)b * 4096 + blockIdx.x) * 2 + 1] = s2[0];
  }
}

__global__ void stats_kernel(const float* __restrict__ part, float* __restrict__ stat) {
  int b = blockIdx.x;
  __shared__ float s1[256], s2[256];
  float a = 0.f, c = 0.f;
  for (int i = threadIdx.x; i < 4096; i += 256) {
    a += part[((long)b * 4096 + i) * 2 + 0];
    c += part[((long)b * 4096 + i) * 2 + 1];
  }
  s1[threadIdx.x] = a; s2[threadIdx.x] = c;
  __syncthreads();
  for (int o = 128; o > 0; o >>= 1) {
    if (threadIdx.x < (unsigned)o) { s1[threadIdx.x] += s1[threadIdx.x + o]; s2[threadIdx.x] += s2[threadIdx.x + o]; }
    __syncthreads();
  }
  if (threadIdx.x == 0) {
    float mu = s1[0] * (1.f / 1048576.f);
    float var = s2[0] * (1.f / 1048576.f) - mu * mu;
    stat[b * 2 + 0] = mu;
    stat[b * 2 + 1] = rsqrtf(var + BN_EPS);
  }
}

__global__ void innorm_kernel(float* __restrict__ xn, const float* __restrict__ stat) {
  const long HW = 1048576;
  int b = blockIdx.y;
  long i = (long)blockIdx.x * 256 + threadIdx.x;
  float mu = stat[b * 2], irs = stat[b * 2 + 1];
  xn[(long)b * HW + i] = (xn[(long)b * HW + i] - mu) * irs;
}

// ---------------------------------------------------------------- skip: avgpool4 + 1x1 conv(1->24)+bias
__global__ void skip_kernel(const float* __restrict__ xn, const float* __restrict__ w,
                            const float* __restrict__ bias, float* __restrict__ out, int B) {
  long t = (long)blockIdx.x * blockDim.x + threadIdx.x;
  long total = (long)B * 256 * 256;
  if (t >= total) return;
  int x = t % 256; long r = t / 256; int y = r % 256; int b = r / 256;
  float s = 0.f;
  for (int dy = 0; dy < 4; ++dy)
    for (int dx = 0; dx < 4; ++dx)
      s += xn[((long)b * 1024 + (y * 4 + dy)) * 1024 + (x * 4 + dx)];
  s *= (1.f / 16.f);
  for (int co = 0; co < 24; ++co)
    out[(((long)b * 24 + co) * 256 + y) * 256 + x] = w[co] * s + bias[co];
}

// ---------------------------------------------------------------- scalar conv + BN + ReLU (small Cin)
__global__ void conv_scalar(const float* __restrict__ in, const float* __restrict__ w,
                            const float* __restrict__ mvec, const float* __restrict__ vvec,
                            float* __restrict__ out, int B, int Cin, int Cout,
                            int inH, int inW, int H, int W, int ks, int stride, int pad) {
  long t = (long)blockIdx.x * blockDim.x + threadIdx.x;
  long total = (long)B * Cout * H * W;
  if (t >= total) return;
  int x = t % W; long r = t / W; int y = r % H; r /= H; int co = r % Cout; int b = r / Cout;
  float s = 0.f;
  for (int ci = 0; ci < Cin; ++ci)
    for (int ky = 0; ky < ks; ++ky) {
      int iy = y * stride + ky - pad; if (iy < 0 || iy >= inH) continue;
      for (int kx = 0; kx < ks; ++kx) {
        int ix = x * stride + kx - pad; if (ix < 0 || ix >= inW) continue;
        s += in[(((long)b * Cin + ci) * inH + iy) * inW + ix] *
             w[((co * Cin + ci) * ks + ky) * ks + kx];
      }
    }
  s = (s - mvec[co]) * rsqrtf(vvec[co] + BN_EPS);
  out[t] = fmaxf(s, 0.f);
}

// ---------------------------------------------------------------- NCHW fp32 -> NHWC f16 (channel-padded)
__global__ void to_nhwc_f16(const float* __restrict__ in, _Float16* __restrict__ out,
                            int B, int Cin, int CP, int H, int W) {
  long t = (long)blockIdx.x * blockDim.x + threadIdx.x;
  long total = (long)B * H * W * CP;
  if (t >= total) return;
  int c = t % CP; long r = t / CP; int x = r % W; r /= W; int y = r % H; int b = r / H;
  float v = (c < Cin) ? in[(((long)b * Cin + c) * H + y) * W + x] : 0.f;
  out[t] = (_Float16)v;
}

// ---------------------------------------------------------------- weight repack: OIHW fp32 -> WMMA B fragments f16
// fragment flat index: (((tap*chunks + ch)*coTiles + coT)*32 + lane)*16 + h
// B-matrix (32x16, f16): lane L -> N = L%16 ; element h -> K = (h<8?h:h+8) + 8*(L>=16)
__global__ void prep_weights(const float* __restrict__ w, _Float16* __restrict__ wf,
                             int Cin, int Cout, int ks, int chunks, int coTiles) {
  long t = (long)blockIdx.x * blockDim.x + threadIdx.x;
  long total = (long)ks * ks * chunks * coTiles * 512;
  if (t >= total) return;
  int h = t & 15;
  int lane = (t >> 4) & 31;
  long rest = t >> 9;
  int coT = rest % coTiles; rest /= coTiles;
  int ch = rest % chunks;
  int tap = rest / chunks;
  int co = coT * 16 + (lane & 15);
  int cin = ch * 32 + (h < 8 ? h : h + 8) + ((lane & 16) >> 1);
  float val = 0.f;
  if (co < Cout && cin < Cin) {
    int ky = tap / ks, kx = tap % ks;
    val = w[(((long)co * Cin + cin) * ks + ky) * ks + kx];
  }
  wf[t] = (_Float16)val;
}

// ---------------------------------------------------------------- WMMA implicit-GEMM conv (NHWC f16 input)
// A (16x32 f16): lane L -> M = L%16 ; element h -> K = (h<8?h:h+8) + 8*(L>=16)
//   NHWC => per-lane K slice is {c0..c0+7, c0+16..c0+23}, two b128 loads
// D (16x16 f32): lane L -> N = L%16 ; vgpr r -> M = r + 8*(L>=16)
// Weights for this block's coT are staged to LDS with async loads.
// epi: 0 = BN+ReLU, 1 = +bias, 2 = +bias then sigmoid
template <int CHUNKS, int KS>
__global__ __launch_bounds__(128) void conv_wmma(
    const _Float16* __restrict__ inF, const _Float16* __restrict__ wf,
    const float* __restrict__ mvec, const float* __restrict__ vvec,
    const float* __restrict__ bias, float* __restrict__ out,
    int B, int Cout, int inH, int inW, int H, int W,
    int stride, int pad, int epi) {
  constexpr int TAPS = KS * KS;
  constexpr int FRAG = TAPS * CHUNKS;
  constexpr int CP = CHUNKS * 32;
  __shared__ _Float16 sw[FRAG * 512];

  int tid = threadIdx.x;
  int lane = tid & 31;
  int wave = tid >> 5;
  int laneHi = lane >> 4;
  int coT = blockIdx.y;
  int coTiles = gridDim.y;

  // ---- stage this coT's weight fragments into LDS
#if HAVE_ASYNC_LDS
  for (int p = tid; p < FRAG * 64; p += 128) {
    int f = p >> 6, q = p & 63;
    const _Float16* g = wf + ((long)f * coTiles + coT) * 512 + q * 8;
    _Float16* l = sw + f * 512 + q * 8;
    __builtin_amdgcn_global_load_async_to_lds_b128((v4i*)g, (lds_v4i_p)l, 0, 0);
  }
  wait_async_all();
#else
  for (int p = tid; p < FRAG * 512; p += 128) {
    int f = p >> 9, q = p & 511;
    sw[p] = wf[((long)f * coTiles + coT) * 512 + q];
  }
#endif
  __syncthreads();

  long Mtot = (long)B * H * W;
  long mBase = ((long)blockIdx.x * 4 + wave) * 16;
  long mRow = mBase + (lane & 15);
  int b = 0, oy = 0, ox = 0;
  bool mvalid = mRow < Mtot;
  if (mvalid) { long r = mRow; ox = (int)(r % W); r /= W; oy = (int)(r % H); b = (int)(r / H); }

  v8f acc = {};
#pragma unroll
  for (int tap = 0; tap < TAPS; ++tap) {
    const int ky = tap / KS, kx = tap % KS;
    int iy = oy * stride + ky - pad;
    int ix = ox * stride + kx - pad;
    bool sp_ok = mvalid && iy >= 0 && iy < inH && ix >= 0 && ix < inW;
    const _Float16* px = inF + (((long)b * inH + iy) * inW + ix) * CP + (laneHi << 3);
#pragma unroll
    for (int ch = 0; ch < CHUNKS; ++ch) {
      v16h afr = {};
      if (sp_ok) {
        v8h lo = *(const v8h*)(px + ch * 32);
        v8h hi = *(const v8h*)(px + ch * 32 + 16);
#pragma unroll
        for (int h = 0; h < 8; ++h) { afr[h] = lo[h]; afr[h + 8] = hi[h]; }
      }
      v16h bfr = *(const v16h*)(sw + ((tap * CHUNKS + ch) * 32 + lane) * 16);
      acc = __builtin_amdgcn_wmma_f32_16x16x32_f16(
          false, afr, false, bfr, (short)0, acc, false, false);
    }
  }

  int co = coT * 16 + (lane & 15);
  if (co < Cout) {
#pragma unroll
    for (int r = 0; r < 8; ++r) {
      long m = mBase + r + (laneHi << 3);
      if (m < Mtot) {
        long rr = m; int x = (int)(rr % W); rr /= W; int y = (int)(rr % H); int bb = (int)(rr / H);
        float val = acc[r];
        if (epi == 0) {
          val = (val - mvec[co]) * rsqrtf(vvec[co] + BN_EPS);
          val = fmaxf(val, 0.f);
        } else {
          val += bias[co];
          if (epi == 2) val = 1.f / (1.f + __expf(-val));
        }
        out[(((long)bb * Cout + co) * H + y) * W + x] = val;
      }
    }
  }
}

// ---------------------------------------------------------------- elementwise add
__global__ void add_kernel(float* __restrict__ a, const float* __restrict__ b, long n) {
  long t = (long)blockIdx.x * blockDim.x + threadIdx.x;
  if (t < n) a[t] += b[t];
}

// ---------------------------------------------------------------- fused = x3 + up(x4) + up(x5)   (C=64 @128^2)
__global__ void fuse_kernel(const float* __restrict__ x3, const float* __restrict__ x4,
                            const float* __restrict__ x5, float* __restrict__ out, int B) {
  long t = (long)blockIdx.x * blockDim.x + threadIdx.x;
  long total = (long)B * 64 * 128 * 128;
  if (t >= total) return;
  int x = t % 128; long r = t / 128; int y = r % 128; r /= 128; int c = r % 64; int b = r / 64;
  float v = x3[t];
  v += bsample(x4 + ((long)b * 64 + c) * 64 * 64, 64, 64, y, x, 128, 128);
  v += bsample(x5 + ((long)b * 64 + c) * 32 * 32, 32, 32, y, x, 128, 128);
  out[t] = v;
}

// ---------------------------------------------------------------- channel L2 norm in place (axis=1)
__global__ void l2norm_ch(float* __restrict__ x, int B, int C, long HW) {
  long t = (long)blockIdx.x * blockDim.x + threadIdx.x;
  long total = (long)B * HW;
  if (t >= total) return;
  long i = t % HW; int b = (int)(t / HW);
  float* p = x + ((long)b * C) * HW + i;
  float ss = 0.f;
  for (int c = 0; c < C; ++c) { float v = p[(long)c * HW]; ss += v * v; }
  float inv = 1.f / fmaxf(sqrtf(ss), 1e-12f);
  for (int c = 0; c < C; ++c) p[(long)c * HW] *= inv;
}

// ---------------------------------------------------------------- pixel unshuffle (B,1,1024,1024)->(B,64,128,128)
__global__ void unshuffle_kernel(const float* __restrict__ xn, float* __restrict__ pu, int B) {
  long t = (long)blockIdx.x * blockDim.x + threadIdx.x;
  long total = (long)B * 64 * 128 * 128;
  if (t >= total) return;
  int x8 = t % 128; long r = t / 128; int y8 = r % 128; r /= 128; int c = r % 64; int b = r / 64;
  int i = c >> 3, j = c & 7;
  pu[t] = xn[((long)b * 1024 + (y8 * 8 + i)) * 1024 + (x8 * 8 + j)];
}

// ---------------------------------------------------------------- softmax over 65 ch -> K1h (B,1,1024,1024)
__global__ void softmax_kernel(const float* __restrict__ K1, float* __restrict__ K1h, int B) {
  long t = (long)blockIdx.x * blockDim.x + threadIdx.x;
  long total = (long)B * 128 * 128;
  if (t >= total) return;
  int x8 = t % 128; long r = t / 128; int y8 = r % 128; int b = r / 128;
  const float* p = K1 + ((long)b * 65) * 16384 + (long)y8 * 128 + x8;
  float mx = -1e30f;
  for (int c = 0; c < 65; ++c) mx = fmaxf(mx, p[(long)c * 16384]);
  float e[64];
  float sum = 0.f;
  for (int c = 0; c < 65; ++c) {
    float ev = __expf(p[(long)c * 16384] - mx);
    sum += ev;
    if (c < 64) e[c] = ev;
  }
  float inv = 1.f / sum;
  float* outp = K1h + (long)b * 1048576;
  for (int c = 0; c < 64; ++c) {
    int Y = y8 * 8 + (c >> 3), X = x8 * 8 + (c & 7);
    outp[(long)Y * 1024 + X] = e[c] * inv;
  }
}

// ---------------------------------------------------------------- scores = K1h * 5x5-peak * up(H1)
__global__ void scores_kernel(const float* __restrict__ K1h, const float* __restrict__ H1,
                              float* __restrict__ out, int B) {
  long t = (long)blockIdx.x * blockDim.x + threadIdx.x;
  long total = (long)B * 1024 * 1024;
  if (t >= total) return;
  int X = t % 1024; long r = t / 1024; int Y = r % 1024; int b = r / 1024;
  const float* plane = K1h + (long)b * 1048576;
  float v = plane[(long)Y * 1024 + X];
  float mx = -1e30f;
  for (int dy = -2; dy <= 2; ++dy) {
    int yy = Y + dy; if (yy < 0 || yy >= 1024) continue;
    for (int dx = -2; dx <= 2; ++dx) {
      int xx = X + dx; if (xx < 0 || xx >= 1024) continue;
      mx = fmaxf(mx, plane[(long)yy * 1024 + xx]);
    }
  }
  float peak = (v == mx && v > 0.05f) ? 1.f : 0.f;
  float h = bsample(H1 + (long)b * 16384, 128, 128, Y, X, 1024, 1024);
  out[t] = v * peak * h;
}

// ---------------------------------------------------------------- top-k via bitonic sort on packed keys
__global__ void pack_keys(const float* __restrict__ scores, unsigned long long* __restrict__ keys,
                          int B, int N) {
  long t = (long)blockIdx.x * blockDim.x + threadIdx.x;
  long total = (long)B * N;
  if (t >= total) return;
  unsigned i = (unsigned)(t % N);
  unsigned so = ford(scores[t]);
  keys[t] = ((unsigned long long)so << 32) | (unsigned long long)(0xFFFFFFFFu - i);
}

__global__ void bitonic_pass(unsigned long long* __restrict__ keys, int N, int k, int j, int B) {
  long t = (long)blockIdx.x * blockDim.x + threadIdx.x;
  long total = (long)B * N;
  if (t >= total) return;
  int b = (int)(t / N);
  int i = (int)(t % N);
  int ixj = i ^ j;
  if (ixj <= i) return;
  unsigned long long* arr = keys + (long)b * N;
  bool descending = ((i & k) == 0);
  unsigned long long a = arr[i], c = arr[ixj];
  bool doswap = descending ? (a < c) : (a > c);
  if (doswap) { arr[i] = c; arr[ixj] = a; }
}

__global__ void topk_extract(const unsigned long long* __restrict__ keys,
                             float* __restrict__ mkpts, float* __restrict__ scores,
                             int B, int N, int K, int W) {
  long t = (long)blockIdx.x * blockDim.x + threadIdx.x;
  long total = (long)B * K;
  if (t >= total) return;
  int b = (int)(t / K);
  unsigned long long key = keys[(long)b * N + (t % K)];
  float sc = fordinv((unsigned)(key >> 32));
  unsigned idx = 0xFFFFFFFFu - (unsigned)(key & 0xFFFFFFFFull);
  mkpts[t * 2 + 0] = (float)(idx % W);
  mkpts[t * 2 + 1] = (float)(idx / W);
  scores[t] = sc;
}

__global__ void num_valid_kernel(const float* __restrict__ scores, int* __restrict__ out, int K) {
  __shared__ int cnt;
  if (threadIdx.x == 0) cnt = 0;
  __syncthreads();
  int local = 0;
  for (int i = threadIdx.x; i < K; i += blockDim.x)
    if (scores[(long)blockIdx.x * K + i] > 0.f) local++;
  atomicAdd(&cnt, local);
  __syncthreads();
  if (threadIdx.x == 0) out[blockIdx.x] = cnt;
}

// ---------------------------------------------------------------- bicubic grid-sample + L2 norm (wave32 / keypoint)
__device__ __forceinline__ void cubw(float f, float* w) {
  const float a = -0.75f;
  float t;
  t = 1.f + f; w[0] = ((a * t - 5.f * a) * t + 8.f * a) * t - 4.f * a;
  t = f;       w[1] = ((a + 2.f) * t - (a + 3.f)) * t * t + 1.f;
  t = 1.f - f; w[2] = ((a + 2.f) * t - (a + 3.f)) * t * t + 1.f;
  t = 2.f - f; w[3] = ((a * t - 5.f * a) * t + 8.f * a) * t - 4.f * a;
}

__global__ void gridsample_kernel(const float* __restrict__ M1, const float* __restrict__ mkpts,
                                  float* __restrict__ featsOut, int B, int K) {
  long wave = ((long)blockIdx.x * blockDim.x + threadIdx.x) >> 5;
  int lane = threadIdx.x & 31;
  long total = (long)B * K;
  if (wave >= total) return;
  int b = (int)(wave / K);
  float posx = mkpts[wave * 2 + 0], posy = mkpts[wave * 2 + 1];
  const int Wd = 128, Hd = 128, Wn = 1024, Hn = 1024;
  float gx = 2.f * posx / (float)(Wn - 1) - 1.f;
  float gy = 2.f * posy / (float)(Hn - 1) - 1.f;
  float px = ((gx + 1.f) * Wd - 1.f) * 0.5f;
  float py = ((gy + 1.f) * Hd - 1.f) * 0.5f;
  float x0 = floorf(px), y0 = floorf(py);
  float fx = px - x0, fy = py - y0;
  float wx[4], wy[4];
  cubw(fx, wx); cubw(fy, wy);
  float acc0 = 0.f, acc1 = 0.f;
  int c0 = lane, c1 = lane + 32;
  const float* base = M1 + ((long)b * 64) * (128 * 128);
  for (int ty = 0; ty < 4; ++ty) {
    int iy = (int)y0 + ty - 1;
    bool oky = iy >= 0 && iy < Hd;
    int iyc = min(max(iy, 0), Hd - 1);
    for (int tx = 0; tx < 4; ++tx) {
      int ix = (int)x0 + tx - 1;
      bool okx = ix >= 0 && ix < Wd;
      int ixc = min(max(ix, 0), Wd - 1);
      float wgt = wy[ty] * wx[tx] * ((oky && okx) ? 1.f : 0.f);
      long off = (long)iyc * Wd + ixc;
      acc0 += wgt * base[(long)c0 * (128 * 128) + off];
      acc1 += wgt * base[(long)c1 * (128 * 128) + off];
    }
  }
  float ss = acc0 * acc0 + acc1 * acc1;
  for (int o = 16; o > 0; o >>= 1) ss += __shfl_xor(ss, o, 32);
  float inv = 1.f / fmaxf(sqrtf(ss), 1e-12f);
  featsOut[wave * 64 + c0] = acc0 * inv;
  featsOut[wave * 64 + c1] = acc1 * inv;
}

// ================================================================ host
extern "C" void kernel_launch(void* const* d_in, const int* in_sizes, int n_in,
                              void* d_out, int out_size, void* d_ws, size_t ws_size,
                              hipStream_t stream) {
  const int B = 4;
  const int TOPK = 4096;
  const float* images = (const float*)d_in[0];

  struct BLW { const float* m, *v, *w; int ci, co, ks; };
  int ii = 1;
  auto takeBL = [&](int ci, int co, int ks) -> BLW {
    BLW r;
    r.m = (const float*)d_in[ii++];
    r.v = (const float*)d_in[ii++];
    r.w = (const float*)d_in[ii++];
    r.ci = ci; r.co = co; r.ks = ks;
    return r;
  };
  BLW b1[4] = { takeBL(1, 4, 3), takeBL(4, 8, 3), takeBL(8, 8, 3), takeBL(8, 24, 3) };
  BLW b2[2] = { takeBL(24, 24, 3), takeBL(24, 24, 3) };
  BLW b3[3] = { takeBL(24, 64, 3), takeBL(64, 64, 3), takeBL(64, 64, 1) };
  BLW b4[3] = { takeBL(64, 64, 3), takeBL(64, 64, 3), takeBL(64, 64, 3) };
  BLW b5[4] = { takeBL(64, 128, 3), takeBL(128, 128, 3), takeBL(128, 128, 3), takeBL(128, 64, 1) };
  const float* fusion_b = (const float*)d_in[ii++];
  BLW fbl[2] = { takeBL(64, 64, 3), takeBL(64, 64, 3) };
  const float* fusion_w = (const float*)d_in[ii++];
  const float* heat_b = (const float*)d_in[ii++];
  BLW hbl[2] = { takeBL(64, 64, 1), takeBL(64, 64, 1) };
  const float* heat_w = (const float*)d_in[ii++];
  const float* kpt_b = (const float*)d_in[ii++];
  BLW kbl[3] = { takeBL(64, 64, 1), takeBL(64, 64, 1), takeBL(64, 64, 1) };
  const float* kpt_w = (const float*)d_in[ii++];
  const float* skip1_b = (const float*)d_in[ii++];
  const float* skip1_w = (const float*)d_in[ii++];

  // ---------------- workspace layout (floats)
  float* WS = (float*)d_ws;
  size_t fo = 0;
  auto FA = [&](size_t n) { float* p = WS + fo; fo += n; return p; };
  float* XN    = FA(4194304);
  float* PART  = FA(32768);
  float* STAT  = FA(16);
  float* BIG1  = FA(16777216);  // b1a ; later PU / kA / kB
  float* BIG2  = FA(8388608);   // b1b ; later K1h ; later sort keys (u64)
  float* BIG3  = FA(8454144);   // b1c ; later scores + K1
  float* X1    = FA(6291456);   // b1d/x1 ; later h1
  float* SKIPB = FA(6291456);   // skip ; later h2
  float* B2A   = FA(6291456);
  float* X2    = FA(6291456);   // x2 ; later f1
  float* B3A   = FA(4194304);   // b3a ; later fused
  float* B3B   = FA(4194304);   // b3b ; later f2
  float* X3    = FA(4194304);
  float* B4A   = FA(1048576);
  float* B4B   = FA(1048576);
  float* X4b   = FA(1048576);
  float* B5A   = FA(524288);
  float* B5B   = FA(524288);
  float* B5C   = FA(524288);
  float* X5b   = FA(262144);
  float* FEATS = FA(4194304);
  float* H1B   = FA(65536);
  _Float16* WH = (_Float16*)FA(1048576);   // repacked f16 weight fragments
  _Float16* F16H = (_Float16*)FA(4194304); // NHWC f16 activation staging (8.4M halves)
  size_t wh_off = 0;

  float* PU = BIG1;
  float* KA = BIG1 + 4194304;
  float* KB = BIG1 + 8388608;
  float* K1H = BIG2;
  unsigned long long* SORTK = (unsigned long long*)BIG2;
  float* SCORES = BIG3;
  float* K1B = BIG3 + 4194304;
  float* H1T1 = X1;
  float* H1T2 = SKIPB;

  float* out_mkpts = (float*)d_out;
  float* out_scores = out_mkpts + (size_t)B * TOPK * 2;
  float* out_feats = out_scores + (size_t)B * TOPK;
  int* out_nv = (int*)(out_feats + (size_t)B * TOPK * 64);

  // ---------------- conv launchers
  auto wconv = [&](const float* in, float* out, const float* w, const float* mm,
                   const float* vv, const float* bias, int ci, int co, int ks,
                   int stride, int pad, int inH, int inW, int H, int Wd, int epi) {
    int chunks = (ci + 31) / 32;
    int coT = (co + 15) / 16;
    long wtotal = (long)ks * ks * chunks * coT * 512;
    prep_weights<<<dim3((unsigned)((wtotal + 255) / 256)), 256, 0, stream>>>(
        w, WH + wh_off, ci, co, ks, chunks, coT);
    long ctotal = (long)B * inH * inW * (chunks * 32);
    to_nhwc_f16<<<dim3((unsigned)((ctotal + 255) / 256)), 256, 0, stream>>>(
        in, F16H, B, ci, chunks * 32, inH, inW);
    long Mtot = (long)B * H * Wd;
    dim3 g((unsigned)((Mtot + 63) / 64), (unsigned)coT);
#define LAUNCH_CONV(CH, KK)                                                          \
    conv_wmma<CH, KK><<<g, 128, 0, stream>>>(F16H, WH + wh_off, mm, vv, bias, out,   \
                                             B, co, inH, inW, H, Wd, stride, pad, epi)
    if (ks == 3) {
      if (chunks == 1) LAUNCH_CONV(1, 3);
      else if (chunks == 2) LAUNCH_CONV(2, 3);
      else LAUNCH_CONV(4, 3);
    } else {
      if (chunks == 2) LAUNCH_CONV(2, 1);
      else if (chunks == 4) LAUNCH_CONV(4, 1);
      else LAUNCH_CONV(1, 1);
    }
#undef LAUNCH_CONV
    wh_off += (size_t)wtotal;
  };
  auto sconv = [&](const float* in, float* out, const BLW& L, int stride, int pad,
                   int inH, int inW, int H, int W) {
    long total = (long)B * L.co * H * W;
    conv_scalar<<<dim3((unsigned)((total + 255) / 256)), 256, 0, stream>>>(
        in, L.w, L.m, L.v, out, B, L.ci, L.co, inH, inW, H, W, L.ks, stride, pad);
  };

  // ---------------- front end
  gray_kernel<<<dim3(4096, B), 256, 0, stream>>>(images, XN, PART);
  stats_kernel<<<dim3(B), 256, 0, stream>>>(PART, STAT);
  innorm_kernel<<<dim3(4096, B), 256, 0, stream>>>(XN, STAT);
  {
    long total = (long)B * 256 * 256;
    skip_kernel<<<dim3((unsigned)((total + 255) / 256)), 256, 0, stream>>>(XN, skip1_w, skip1_b, SKIPB, B);
  }

  // block1 (scalar convs, small Cin)
  sconv(XN,   BIG1, b1[0], 1, 1, 1024, 1024, 1024, 1024);
  sconv(BIG1, BIG2, b1[1], 2, 1, 1024, 1024, 512, 512);
  sconv(BIG2, BIG3, b1[2], 1, 1, 512, 512, 512, 512);
  sconv(BIG3, X1,   b1[3], 2, 1, 512, 512, 256, 256);

  add_kernel<<<dim3((unsigned)((6291456 + 255) / 256)), 256, 0, stream>>>(X1, SKIPB, 6291456);

  // block2 (WMMA)
  wconv(X1,  B2A, b2[0].w, b2[0].m, b2[0].v, nullptr, 24, 24, 3, 1, 1, 256, 256, 256, 256, 0);
  wconv(B2A, X2,  b2[1].w, b2[1].m, b2[1].v, nullptr, 24, 24, 3, 1, 1, 256, 256, 256, 256, 0);
  // block3
  wconv(X2,  B3A, b3[0].w, b3[0].m, b3[0].v, nullptr, 24, 64, 3, 2, 1, 256, 256, 128, 128, 0);
  wconv(B3A, B3B, b3[1].w, b3[1].m, b3[1].v, nullptr, 64, 64, 3, 1, 1, 128, 128, 128, 128, 0);
  wconv(B3B, X3,  b3[2].w, b3[2].m, b3[2].v, nullptr, 64, 64, 1, 1, 0, 128, 128, 128, 128, 0);
  // block4
  wconv(X3,  B4A, b4[0].w, b4[0].m, b4[0].v, nullptr, 64, 64, 3, 2, 1, 128, 128, 64, 64, 0);
  wconv(B4A, B4B, b4[1].w, b4[1].m, b4[1].v, nullptr, 64, 64, 3, 1, 1, 64, 64, 64, 64, 0);
  wconv(B4B, X4b, b4[2].w, b4[2].m, b4[2].v, nullptr, 64, 64, 3, 1, 1, 64, 64, 64, 64, 0);
  // block5
  wconv(X4b, B5A, b5[0].w, b5[0].m, b5[0].v, nullptr, 64, 128, 3, 2, 1, 64, 64, 32, 32, 0);
  wconv(B5A, B5B, b5[1].w, b5[1].m, b5[1].v, nullptr, 128, 128, 3, 1, 1, 32, 32, 32, 32, 0);
  wconv(B5B, B5C, b5[2].w, b5[2].m, b5[2].v, nullptr, 128, 128, 3, 1, 1, 32, 32, 32, 32, 0);
  wconv(B5C, X5b, b5[3].w, b5[3].m, b5[3].v, nullptr, 128, 64, 1, 1, 0, 32, 32, 32, 32, 0);

  // fusion
  {
    long total = (long)B * 64 * 128 * 128;
    fuse_kernel<<<dim3((unsigned)((total + 255) / 256)), 256, 0, stream>>>(X3, X4b, X5b, B3A, B);
  }
  wconv(B3A, X2,  fbl[0].w, fbl[0].m, fbl[0].v, nullptr, 64, 64, 3, 1, 1, 128, 128, 128, 128, 0);
  wconv(X2,  B3B, fbl[1].w, fbl[1].m, fbl[1].v, nullptr, 64, 64, 3, 1, 1, 128, 128, 128, 128, 0);
  wconv(B3B, FEATS, fusion_w, nullptr, nullptr, fusion_b, 64, 64, 1, 1, 0, 128, 128, 128, 128, 1);

  // heat head (uses unnormalized feats)
  wconv(FEATS, H1T1, hbl[0].w, hbl[0].m, hbl[0].v, nullptr, 64, 64, 1, 1, 0, 128, 128, 128, 128, 0);
  wconv(H1T1,  H1T2, hbl[1].w, hbl[1].m, hbl[1].v, nullptr, 64, 64, 1, 1, 0, 128, 128, 128, 128, 0);
  wconv(H1T2,  H1B,  heat_w, nullptr, nullptr, heat_b, 64, 1, 1, 1, 0, 128, 128, 128, 128, 2);

  // M1 = l2norm(feats, axis=1) in place
  {
    long total = (long)B * 128 * 128;
    l2norm_ch<<<dim3((unsigned)((total + 255) / 256)), 256, 0, stream>>>(FEATS, B, 64, 128 * 128);
  }

  // keypoint head
  {
    long total = (long)B * 64 * 128 * 128;
    unshuffle_kernel<<<dim3((unsigned)((total + 255) / 256)), 256, 0, stream>>>(XN, PU, B);
  }
  wconv(PU, KA, kbl[0].w, kbl[0].m, kbl[0].v, nullptr, 64, 64, 1, 1, 0, 128, 128, 128, 128, 0);
  wconv(KA, KB, kbl[1].w, kbl[1].m, kbl[1].v, nullptr, 64, 64, 1, 1, 0, 128, 128, 128, 128, 0);
  wconv(KB, KA, kbl[2].w, kbl[2].m, kbl[2].v, nullptr, 64, 64, 1, 1, 0, 128, 128, 128, 128, 0);
  wconv(KA, K1B, kpt_w, nullptr, nullptr, kpt_b, 64, 65, 1, 1, 0, 128, 128, 128, 128, 1);

  {
    long total = (long)B * 128 * 128;
    softmax_kernel<<<dim3((unsigned)((total + 255) / 256)), 256, 0, stream>>>(K1B, K1H, B);
  }
  {
    long total = (long)B * 1024 * 1024;
    scores_kernel<<<dim3((unsigned)((total + 255) / 256)), 256, 0, stream>>>(K1H, H1B, SCORES, B);
  }

  // top-k via full bitonic sort (descending), N = 2^20 per batch
  const int N = 1024 * 1024;
  {
    long total = (long)B * N;
    pack_keys<<<dim3((unsigned)((total + 255) / 256)), 256, 0, stream>>>(SCORES, SORTK, B, N);
    unsigned blocks = (unsigned)((total + 255) / 256);
    for (int k = 2; k <= N; k <<= 1)
      for (int j = k >> 1; j > 0; j >>= 1)
        bitonic_pass<<<dim3(blocks), 256, 0, stream>>>(SORTK, N, k, j, B);
  }
  {
    long total = (long)B * TOPK;
    topk_extract<<<dim3((unsigned)((total + 255) / 256)), 256, 0, stream>>>(
        SORTK, out_mkpts, out_scores, B, N, TOPK, 1024);
  }
  num_valid_kernel<<<dim3(B), 256, 0, stream>>>(out_scores, out_nv, TOPK);

  {
    long threads = (long)B * TOPK * 32;
    gridsample_kernel<<<dim3((unsigned)((threads + 255) / 256)), 256, 0, stream>>>(
        FEATS, out_mkpts, out_feats, B, TOPK);
  }

  (void)in_sizes; (void)n_in; (void)out_size; (void)ws_size;
}
